// RSSMCore_76003741270410
// MI455X (gfx1250) — matrix operand order repaired and checked by reading, not compiled, not run on hardware
//
#include <hip/hip_runtime.h>
#include <hip/hip_bf16.h>

// MI455X (gfx1250) RSSM core: bf16 WMMA GEMMs (f32 accumulate), weights L2-resident,
// per-block weight tiles staged in LDS via async global->LDS copies (ASYNCcnt),
// double-buffered against WMMA compute.

typedef __attribute__((ext_vector_type(16))) __bf16 v16bf;
typedef __attribute__((ext_vector_type(8)))  __bf16 v8bf;
typedef __attribute__((ext_vector_type(8)))  float  v8f;
typedef __attribute__((ext_vector_type(4)))  float  v4f;
typedef __attribute__((ext_vector_type(4)))  int    v4i;

#define T_STEPS 64
#define BATCH   1024
#define EMBED_D 1024
#define ACT_D   32
#define DETER_D 2048
#define STOCH_D 64
#define HID_D   1024
#define GLOB_D  64
#define MIN_STD 0.1f

// B-tile staging: 128 rows x 32 K (bf16) = 64B data per row, padded to 80B so the
// 16 fragment rows map to 16 distinct LDS banks; 16B aligned for b128 ds loads.
#define LDS_ROW 80
#define LDS_TILE (128 * LDS_ROW)

#if __has_builtin(__builtin_amdgcn_global_load_async_to_lds_b128) && \
    __has_builtin(__builtin_amdgcn_s_wait_asynccnt)
#define USE_ASYNC_LDS 1
#endif

// Address-space qualified v4i pointers (per the builtin's signature: AS1 src, AS3 dst).
typedef __attribute__((address_space(1))) v4i gv4i;
typedef __attribute__((address_space(3))) v4i lv4i;

// ---------------- WMMA helpers ----------------

__device__ __forceinline__ v8f wmma_bf16(v16bf a, v16bf b, v8f c) {
    return __builtin_amdgcn_wmma_f32_16x16x32_bf16(
        /*neg_a=*/false, a, /*neg_b=*/false, b,
        /*c_mod=*/(short)0, c, /*reuse_a=*/false, /*reuse_b=*/false);
}

// A fragment 16(M)x32(K), bf16 row-major source with leading dim ld.
__device__ __forceinline__ v16bf load_a_bf16(const __bf16* __restrict__ base, int ld, bool kill) {
    const int lane = threadIdx.x & 31;
    const int m = lane & 15, kh = lane >> 4;
    const __bf16* p = base + (size_t)m * ld + 8 * kh;
    v8bf lo = *(const v8bf*)p;
    v8bf hi = *(const v8bf*)(p + 16);
    v16bf r;
#pragma unroll
    for (int i = 0; i < 8; ++i) { r[i] = lo[i]; r[i + 8] = hi[i]; }
    if (kill) {
#pragma unroll
        for (int i = 0; i < 16; ++i) r[i] = (__bf16)0.0f;
    }
    return r;
}

// Same A fragment from an f32 row-major source (convert in-register).
__device__ __forceinline__ v16bf load_a_f32(const float* __restrict__ base, int ld) {
    const int lane = threadIdx.x & 31;
    const int m = lane & 15, kh = lane >> 4;
    const float* p = base + (size_t)m * ld + 8 * kh;
    v4f a0 = *(const v4f*)p;
    v4f a1 = *(const v4f*)(p + 4);
    v4f b0 = *(const v4f*)(p + 16);
    v4f b1 = *(const v4f*)(p + 20);
    v16bf r;
#pragma unroll
    for (int i = 0; i < 4; ++i) {
        r[i]      = (__bf16)a0[i];
        r[i + 4]  = (__bf16)a1[i];
        r[i + 8]  = (__bf16)b0[i];
        r[i + 12] = (__bf16)b1[i];
    }
    return r;
}

// B fragment 32(K)x16(N) read from the LDS-staged tile (rows = N, 64B K-data/row).
__device__ __forceinline__ v16bf load_b_lds(const unsigned char* buf, int nrow) {
    const int lane = threadIdx.x & 31;
    const int n = lane & 15, kh = lane >> 4;
    const unsigned char* p = buf + (size_t)(nrow + n) * LDS_ROW + 16 * kh;
    v8bf x0 = *(const v8bf*)p;
    v8bf x1 = *(const v8bf*)(p + 32);
    v16bf r;
#pragma unroll
    for (int i = 0; i < 8; ++i) { r[i] = x0[i]; r[i + 8] = x1[i]; }
    return r;
}

// Copy one 16B chunk global -> LDS (async on gfx1250; sync fallback otherwise).
__device__ __forceinline__ void cp16_g2l(const __bf16* g, unsigned char* l) {
#ifdef USE_ASYNC_LDS
    __builtin_amdgcn_global_load_async_to_lds_b128(
        (gv4i*)(void*)g, (lv4i*)(void*)l, 0, 0);
#else
    *(v8bf*)l = *(const v8bf*)g;
#endif
}

__device__ __forceinline__ void wait_async_le2() {
#ifdef USE_ASYNC_LDS
    __builtin_amdgcn_s_wait_asynccnt(2);
#endif
}
__device__ __forceinline__ void wait_async_le0() {
#ifdef USE_ASYNC_LDS
    __builtin_amdgcn_s_wait_asynccnt(0);
#endif
}

// Stage W rows [nblk, nblk+128), cols [kb, kb+32) into buf. 256 threads, 2 chunks each.
__device__ __forceinline__ void fill_tile(unsigned char* buf, const __bf16* __restrict__ W,
                                          int nblk, int K, int kb) {
    const int tid = threadIdx.x;
#pragma unroll
    for (int i = 0; i < 2; ++i) {
        const int c = tid + i * 256;        // 512 chunks of 16B
        const int row = c >> 2, q = c & 3;
        const __bf16* g = W + (size_t)(nblk + row) * K + kb + q * 8;
        unsigned char* l = buf + (size_t)row * LDS_ROW + q * 16;
        cp16_g2l(g, l);
    }
}

__device__ __forceinline__ float elu_f(float x)  { return x > 0.0f ? x : (__expf(x) - 1.0f); }
__device__ __forceinline__ float sigm_f(float x) { return 1.0f / (1.0f + __expf(-x)); }
__device__ __forceinline__ float softplus_f(float x) {
    return fmaxf(x, 0.0f) + log1pf(__expf(-fabsf(x)));
}

// ---------------- Generic bf16 WMMA GEMM ----------------
// C[M,N] = concat_K(A1_bf16[M,K1], A2_f32[M,K2]) @ W[N, K1+K2]^T (+bias)
// Block: 256 thr = 8 waves, block tile 128(M) x 128(N); wave tile 32x64.
// Weight tile double-buffered in LDS via async copies; A direct from global.
// mode 0: Cf = f32 result. mode 1: Cb = bf16(elu(result)).
__global__ void __launch_bounds__(256) gemm_bf16_wmma(
    const __bf16* __restrict__ A1, int lda1, int K1,
    const float*  __restrict__ A2, int lda2, int K2,
    const unsigned char* __restrict__ rmask,
    const __bf16* __restrict__ W,
    const float* __restrict__ bias,
    float* __restrict__ Cf, __bf16* __restrict__ Cb,
    int ldc, int mode)
{
    __shared__ __align__(16) unsigned char smem[2 * LDS_TILE];

    const int lane  = threadIdx.x & 31;
    const int wave  = threadIdx.x >> 5;
    const int m0    = blockIdx.y * 128 + (wave & 3) * 32;   // wave M base (2 subtiles)
    const int nwav  = (wave >> 2) * 64;                      // wave N base within block tile
    const int nblk  = blockIdx.x * 128;
    const int K     = K1 + K2;
    const int mlane = lane & 15;
    const int mh    = lane >> 4;

    bool kill0 = false, kill1 = false;
    if (rmask) {
        kill0 = rmask[m0 + mlane] != 0;
        kill1 = rmask[m0 + 16 + mlane] != 0;
    }

    v8f acc[2][4] = {};

    // Prologue: start filling buffer 0 with the first weight tile.
    fill_tile(smem, W, nblk, K, 0);

    int it = 0;
    for (int kb = 0; kb < K; kb += 32, ++it) {
        unsigned char* bufc = smem + (size_t)(it & 1) * LDS_TILE;
        unsigned char* bufn = smem + (size_t)((it & 1) ^ 1) * LDS_TILE;
        const bool more = (kb + 32) < K;

        if (more) {
            fill_tile(bufn, W, nblk, K, kb + 32);   // overlap copy of next tile
            wait_async_le2();                        // current tile's 2 ops complete (in-order)
        } else {
            wait_async_le0();
        }
        __syncthreads();                             // whole tile visible to all waves

        v16bf afrag[2];
        if (kb < K1) {
            afrag[0] = load_a_bf16(A1 + (size_t)m0 * lda1 + kb, lda1, kill0);
            afrag[1] = load_a_bf16(A1 + (size_t)(m0 + 16) * lda1 + kb, lda1, kill1);
        } else {
            const int k2 = kb - K1;
            afrag[0] = load_a_f32(A2 + (size_t)m0 * lda2 + k2, lda2);
            afrag[1] = load_a_f32(A2 + (size_t)(m0 + 16) * lda2 + k2, lda2);
        }
        v16bf bfrag[4];
#pragma unroll
        for (int j = 0; j < 4; ++j)
            bfrag[j] = load_b_lds(bufc, nwav + 16 * j);
#pragma unroll
        for (int i = 0; i < 2; ++i)
#pragma unroll
            for (int j = 0; j < 4; ++j)
                acc[i][j] = wmma_bf16(afrag[i], bfrag[j], acc[i][j]);

        __syncthreads();                             // reads of bufc done before its refill
    }

    // Epilogue. C layout: lanes 0-15 -> N=lane, M=vgpr; lanes 16-31 -> N=lane-16, M=8+vgpr.
#pragma unroll
    for (int i = 0; i < 2; ++i) {
#pragma unroll
        for (int j = 0; j < 4; ++j) {
            const int nn = nblk + nwav + 16 * j + mlane;
            const float bv = bias ? bias[nn] : 0.0f;
#pragma unroll
            for (int r = 0; r < 8; ++r) {
                const int mm = m0 + 16 * i + 8 * mh + r;
                float v = acc[i][j][r] + bv;
                if (mode) {
                    Cb[(size_t)mm * ldc + nn] = (__bf16)elu_f(v);
                } else {
                    Cf[(size_t)mm * ldc + nn] = v;
                }
            }
        }
    }
}

// ---------------- Elementwise kernels ----------------

__global__ void gru_combine(const float* __restrict__ Gi, const float* __restrict__ Gh,
                            __bf16* __restrict__ h_bf,
                            const unsigned char* __restrict__ reset,
                            float* __restrict__ states_t)
{
    const int idx = blockIdx.x * blockDim.x + threadIdx.x;   // B * DETER
    const int b = idx >> 11;
    const int n = idx & 2047;
    const size_t gb = (size_t)b * 6144;
    float gir = Gi[gb + n],        ghr = Gh[gb + n];
    float giz = Gi[gb + 2048 + n], ghz = Gh[gb + 2048 + n];
    float gin = Gi[gb + 4096 + n], ghn = Gh[gb + 4096 + n];
    float hprev = (float)h_bf[(size_t)b * 2048 + n];
    if (reset[b]) hprev = 0.0f;
    const float rg = sigm_f(gir + ghr);
    const float ug = sigm_f(giz + ghz);
    const float ng = tanhf(gin + rg * ghn);
    const float hn = (1.0f - ug) * ng + ug * hprev;
    states_t[(size_t)b * 2176 + n] = hn;
    h_bf[(size_t)b * 2048 + n] = (__bf16)hn;
}

__global__ void head_out(const float* __restrict__ Pp, const float* __restrict__ Pq,
                         const float* __restrict__ noise_t, const float* __restrict__ glob,
                         float* __restrict__ priors_t, float* __restrict__ posts_t,
                         float* __restrict__ states_t, __bf16* __restrict__ z_bf)
{
    const int idx = blockIdx.x * blockDim.x + threadIdx.x;   // B * STOCH
    const int b = idx >> 6;
    const int s = idx & 63;
    const size_t pb = (size_t)b * 128;
    const float pm = Pp[pb + s];
    const float ps = softplus_f(Pp[pb + 64 + s]) + MIN_STD;
    priors_t[pb + s] = pm;
    priors_t[pb + 64 + s] = ps;
    const float qm = Pq[pb + s];
    const float qs = softplus_f(Pq[pb + 64 + s]) + MIN_STD;
    posts_t[pb + s] = qm;
    posts_t[pb + 64 + s] = qs;
    const float sample = qm + qs * noise_t[(size_t)b * 64 + s];
    states_t[(size_t)b * 2176 + 2048 + s] = sample;
    states_t[(size_t)b * 2176 + 2112 + s] = glob[(size_t)b * 64 + s];
    z_bf[(size_t)b * 64 + s] = (__bf16)sample;
}

// ---------------- Prepass kernels ----------------

__global__ void cvt_bf16(__bf16* __restrict__ dst, const float* __restrict__ src, long n) {
    const long i = (long)blockIdx.x * blockDim.x + threadIdx.x;
    if (i < n) dst[i] = (__bf16)src[i];
}

// dst (rows x (ka+kb)) = concat columns of a (rows x ka) and b (rows x kb)
__global__ void pack2_bf16(__bf16* __restrict__ dst,
                           const float* __restrict__ a, int ka,
                           const float* __restrict__ b, int kb, long total) {
    const long i = (long)blockIdx.x * blockDim.x + threadIdx.x;
    if (i >= total) return;
    const int kt = ka + kb;
    const long r = i / kt;
    const int  c = (int)(i % kt);
    dst[i] = (__bf16)(c < ka ? a[r * (long)ka + c] : b[r * (long)kb + (c - ka)]);
}

__global__ void init_state_k(__bf16* __restrict__ h_bf, __bf16* __restrict__ z_bf,
                             const float* __restrict__ in_state) {
    const int idx = blockIdx.x * blockDim.x + threadIdx.x;   // B * 2112
    const int b = idx / 2112;
    const int n = idx % 2112;
    const float v = in_state[(size_t)b * 2176 + n];
    if (n < 2048) h_bf[(size_t)b * 2048 + n] = (__bf16)v;
    else          z_bf[(size_t)b * 64 + (n - 2048)] = (__bf16)v;
}

// ---------------- Launch ----------------

extern "C" void kernel_launch(void* const* d_in, const int* in_sizes, int n_in,
                              void* d_out, int out_size, void* d_ws, size_t ws_size,
                              hipStream_t stream) {
    (void)in_sizes; (void)n_in; (void)out_size; (void)ws_size;

    const float* embed    = (const float*)d_in[0];
    const float* action   = (const float*)d_in[1];
    const unsigned char* reset = (const unsigned char*)d_in[2];
    const float* in_state = (const float*)d_in[3];
    const float* glob     = (const float*)d_in[4];
    const float* noise    = (const float*)d_in[5];
    const float* z_mlp_w  = (const float*)d_in[6];
    const float* z_mlp_b  = (const float*)d_in[7];
    const float* a_mlp_w  = (const float*)d_in[8];
    const float* w_ih     = (const float*)d_in[9];
    const float* w_hh     = (const float*)d_in[10];
    const float* b_ih     = (const float*)d_in[11];
    const float* b_hh     = (const float*)d_in[12];
    const float* ph_w     = (const float*)d_in[13];
    const float* ph_b     = (const float*)d_in[14];
    const float* pr_w     = (const float*)d_in[15];
    const float* pr_b     = (const float*)d_in[16];
    const float* qh_w     = (const float*)d_in[17];
    const float* qh_b     = (const float*)d_in[18];
    const float* qe_w     = (const float*)d_in[19];
    const float* po_w     = (const float*)d_in[20];
    const float* po_b     = (const float*)d_in[21];

    char* ws = (char*)d_ws;
    size_t off = 0;
    auto take = [&](size_t bytes) -> char* {
        char* p = ws + off;
        off = (off + bytes + 255) & ~(size_t)255;
        return p;
    };

    __bf16* W_za  = (__bf16*)take((size_t)1024 * 96 * 2);     // [Wz | Wa], 1024 x 96
    __bf16* W_ih  = (__bf16*)take((size_t)6144 * 1024 * 2);
    __bf16* W_hh  = (__bf16*)take((size_t)6144 * 2048 * 2);
    __bf16* W_ph  = (__bf16*)take((size_t)1024 * 2048 * 2);
    __bf16* W_pr  = (__bf16*)take((size_t)128 * 1024 * 2);
    __bf16* W_qhe = (__bf16*)take((size_t)1024 * 3072 * 2);   // [W_post_h | W_post_e]
    __bf16* W_po  = (__bf16*)take((size_t)128 * 1024 * 2);
    __bf16* h_bf  = (__bf16*)take((size_t)BATCH * 2048 * 2);
    __bf16* z_bf  = (__bf16*)take((size_t)BATCH * 64 * 2);
    __bf16* za_bf = (__bf16*)take((size_t)BATCH * 1024 * 2);
    __bf16* pri_in = (__bf16*)take((size_t)BATCH * 1024 * 2);
    __bf16* pos_in = (__bf16*)take((size_t)BATCH * 1024 * 2);
    float*  Gi    = (float*)take((size_t)BATCH * 6144 * 4);
    float*  Gh    = (float*)take((size_t)BATCH * 6144 * 4);
    float*  Pp    = (float*)take((size_t)BATCH * 128 * 4);
    float*  Pq    = (float*)take((size_t)BATCH * 128 * 4);

    const int thr = 256;
    auto nb = [&](long n) { return (unsigned)((n + thr - 1) / thr); };

    // Prepass: weight conversion / packing (every call; deterministic).
    pack2_bf16<<<nb((long)1024 * 96), thr, 0, stream>>>(W_za, z_mlp_w, 64, a_mlp_w, 32, (long)1024 * 96);
    cvt_bf16<<<nb((long)6144 * 1024), thr, 0, stream>>>(W_ih, w_ih, (long)6144 * 1024);
    cvt_bf16<<<nb((long)6144 * 2048), thr, 0, stream>>>(W_hh, w_hh, (long)6144 * 2048);
    cvt_bf16<<<nb((long)1024 * 2048), thr, 0, stream>>>(W_ph, ph_w, (long)1024 * 2048);
    cvt_bf16<<<nb((long)128 * 1024), thr, 0, stream>>>(W_pr, pr_w, (long)128 * 1024);
    pack2_bf16<<<nb((long)1024 * 3072), thr, 0, stream>>>(W_qhe, qh_w, 2048, qe_w, 1024, (long)1024 * 3072);
    cvt_bf16<<<nb((long)128 * 1024), thr, 0, stream>>>(W_po, po_w, (long)128 * 1024);
    init_state_k<<<nb((long)BATCH * 2112), thr, 0, stream>>>(h_bf, z_bf, in_state);

    float* priors = (float*)d_out;
    float* posts  = priors + (size_t)T_STEPS * BATCH * 128;
    float* states = posts + (size_t)T_STEPS * BATCH * 128;

    const dim3 blk(256);
    for (int t = 0; t < T_STEPS; ++t) {
        const unsigned char* rs = reset + (size_t)t * BATCH;
        const float* act_t = action + (size_t)t * BATCH * ACT_D;
        const float* emb_t = embed  + (size_t)t * BATCH * EMBED_D;
        const float* noi_t = noise  + (size_t)t * BATCH * STOCH_D;
        float* st_t  = states + (size_t)t * BATCH * 2176;
        float* pri_t = priors + (size_t)t * BATCH * 128;
        float* pos_t = posts  + (size_t)t * BATCH * 128;

        // za = elu([z, a] @ W_za^T + z_b)  (z reset-masked; a from f32 input)
        gemm_bf16_wmma<<<dim3(1024 / 128, BATCH / 128), blk, 0, stream>>>(
            z_bf, 64, 64, act_t, 32, 32, rs, W_za, z_mlp_b, nullptr, za_bf, 1024, 1);
        // Gi = za @ W_ih^T + b_ih
        gemm_bf16_wmma<<<dim3(6144 / 128, BATCH / 128), blk, 0, stream>>>(
            za_bf, 1024, 1024, nullptr, 0, 0, nullptr, W_ih, b_ih, Gi, nullptr, 6144, 0);
        // Gh = (h * mask) @ W_hh^T + b_hh
        gemm_bf16_wmma<<<dim3(6144 / 128, BATCH / 128), blk, 0, stream>>>(
            h_bf, 2048, 2048, nullptr, 0, 0, rs, W_hh, b_hh, Gh, nullptr, 6144, 0);
        // GRU combine -> h_new (f32 into states[t][:, :2048], bf16 into h_bf)
        gru_combine<<<nb((long)BATCH * 2048), thr, 0, stream>>>(Gi, Gh, h_bf, rs, st_t);
        // prior_in = elu(h_new @ W_ph^T + b)
        gemm_bf16_wmma<<<dim3(1024 / 128, BATCH / 128), blk, 0, stream>>>(
            h_bf, 2048, 2048, nullptr, 0, 0, nullptr, W_ph, ph_b, nullptr, pri_in, 1024, 1);
        // post_in = elu([h_new, e] @ [W_qh | W_qe]^T + b)  (e converted f32->bf16 in-register)
        gemm_bf16_wmma<<<dim3(1024 / 128, BATCH / 128), blk, 0, stream>>>(
            h_bf, 2048, 2048, emb_t, 1024, 1024, nullptr, W_qhe, qh_b, nullptr, pos_in, 1024, 1);
        // prior head (N=128)
        gemm_bf16_wmma<<<dim3(1, BATCH / 128), blk, 0, stream>>>(
            pri_in, 1024, 1024, nullptr, 0, 0, nullptr, W_pr, pr_b, Pp, nullptr, 128, 0);
        // post head (N=128)
        gemm_bf16_wmma<<<dim3(1, BATCH / 128), blk, 0, stream>>>(
            pos_in, 1024, 1024, nullptr, 0, 0, nullptr, W_po, po_b, Pq, nullptr, 128, 0);
        // to_mean_std, sample, write priors/posts/states tail, z_bf for next step
        head_out<<<nb((long)BATCH * STOCH_D), thr, 0, stream>>>(
            Pp, Pq, noi_t, glob, pri_t, pos_t, st_t, z_bf);
    }
}
